// Decoder_25821343384352
// MI455X (gfx1250) — matrix operand order repaired and checked by reading, not compiled, not run on hardware
//
#include <hip/hip_runtime.h>

// ---------------------------------------------------------------------------
// MI455X (gfx1250) decoder: bf16 WMMA GEMMs + WMMA flash attention.
// Residual stream fp32; all matmuls v_wmma_f32_16x16x32_bf16 (f32 accum).
// GEMM tiles staged with async global->LDS copies (ASYNCcnt) when available,
// double-buffered so the async copy overlaps the matrix pipe.
// ---------------------------------------------------------------------------

typedef __bf16 bf16_t;
typedef __bf16 v16bf __attribute__((ext_vector_type(16)));
typedef __bf16 v8bf  __attribute__((ext_vector_type(8)));
typedef float  v8f   __attribute__((ext_vector_type(8)));
typedef int    v4i   __attribute__((ext_vector_type(4)));

#define B_    2
#define S_    2048
#define D_    1024
#define KSEG  256
#define LAYERS 2
#define H_    16
#define HD_   64
#define F_    4096
#define TOK   (B_ * S_)   // 4096 tokens

#if defined(__HIP_DEVICE_COMPILE__) && \
    __has_builtin(__builtin_amdgcn_global_load_async_to_lds_b128)
#define ASYNC_COPY 1
#else
#define ASYNC_COPY 0
#endif

static __device__ __forceinline__ void cp_b128(const bf16_t* g, __bf16* l) {
#if ASYNC_COPY
  __builtin_amdgcn_global_load_async_to_lds_b128(
      (__attribute__((address_space(1))) v4i*)g,
      (__attribute__((address_space(3))) v4i*)l, 0, 0);
#else
  *(v8bf*)l = *(const v8bf*)g;
#endif
}

static __device__ __forceinline__ void async_wait_all() {
#if ASYNC_COPY
#if __has_builtin(__builtin_amdgcn_s_wait_asynccnt)
  __builtin_amdgcn_s_wait_asynccnt(0);
#else
  asm volatile("s_wait_asynccnt 0x0" ::: "memory");
#endif
#endif
}

static __device__ __forceinline__ v8f v8f_zero() {
  v8f z;
#pragma unroll
  for (int i = 0; i < 8; ++i) z[i] = 0.f;
  return z;
}

static __device__ __forceinline__ v16bf cat8(v8bf lo, v8bf hi) {
  return __builtin_shufflevector(lo, hi, 0,1,2,3,4,5,6,7,8,9,10,11,12,13,14,15);
}

static __device__ __forceinline__ v8f wmma_bf16(v16bf a, v16bf b, v8f c) {
  // (neg_a, A, neg_b, B, c_mod, C, reuse_a, reuse_b)
  return __builtin_amdgcn_wmma_f32_16x16x32_bf16(false, a, false, b, (short)0, c,
                                                 false, false);
}

// ---------------------------------------------------------------------------
// fp32 -> bf16 conversion (weights)
// ---------------------------------------------------------------------------
__global__ void cvt_f32_bf16(const float* __restrict__ s, bf16_t* __restrict__ d,
                             long long n) {
  long long i = (long long)blockIdx.x * blockDim.x + threadIdx.x;
  if (i < n) d[i] = (bf16_t)s[i];
}

// ---------------------------------------------------------------------------
// Ragged expand: seg[b,t] = searchsorted(boundaries[b], t, 'left')
// ---------------------------------------------------------------------------
__global__ void seg_kernel(const int* __restrict__ boundaries,
                           const int* __restrict__ counts,
                           int* __restrict__ seg) {
  int t = blockIdx.x * blockDim.x + threadIdx.x;
  if (t >= TOK) return;
  int b = t / S_;
  int pos = t % S_;
  const int* bnd = boundaries + b * KSEG;
  int cnt = counts[b];
  int lo = 0, hi = KSEG;
  while (lo < hi) {
    int mid = (lo + hi) >> 1;
    int v = (mid < cnt) ? bnd[mid] : S_;
    if (v < pos) lo = mid + 1; else hi = mid;
  }
  seg[t] = lo;
}

__global__ void gather_kernel(const float* __restrict__ xp,
                              const float* __restrict__ xr,
                              const float* __restrict__ start_emb,
                              const int* __restrict__ seg,
                              float* __restrict__ x) {
  int idx = blockIdx.x * blockDim.x + threadIdx.x;   // over TOK * (D/4)
  if (idx >= TOK * (D_ / 4)) return;
  int d4 = idx % (D_ / 4);
  int t  = idx / (D_ / 4);
  int b  = t / S_;
  int sg = seg[t];
  float4 r = ((const float4*)xr)[idx];
  float4 e = (sg == 0) ? ((const float4*)start_emb)[d4]
                       : ((const float4*)xp)[(size_t)(b * KSEG + sg - 1) * (D_ / 4) + d4];
  float4 o = make_float4(r.x + e.x, r.y + e.y, r.z + e.z, r.w + e.w);
  ((float4*)x)[idx] = o;
}

// ---------------------------------------------------------------------------
// RMSNorm: one block per row, 256 threads, D=1024 (one float4 per thread)
// ---------------------------------------------------------------------------
template <bool OUTBF>
__global__ __launch_bounds__(256) void rmsnorm_k(const float* __restrict__ x,
                                                 const float* __restrict__ w,
                                                 void* __restrict__ out) {
  int row = blockIdx.x;
  int tid = threadIdx.x;
  const float4* xr = (const float4*)(x + (size_t)row * D_);
  float4 v = xr[tid];
  float ss = v.x * v.x + v.y * v.y + v.z * v.z + v.w * v.w;
  __shared__ float red[256];
  red[tid] = ss;
  __syncthreads();
  for (int s = 128; s > 0; s >>= 1) {
    if (tid < s) red[tid] += red[tid + s];
    __syncthreads();
  }
  float r = rsqrtf(red[0] / (float)D_ + 1e-5f);
  float4 wv = ((const float4*)w)[tid];
  float o0 = v.x * r * wv.x, o1 = v.y * r * wv.y;
  float o2 = v.z * r * wv.z, o3 = v.w * r * wv.w;
  if (OUTBF) {
    bf16_t* o = (bf16_t*)out + (size_t)row * D_ + tid * 4;
    o[0] = (bf16_t)o0; o[1] = (bf16_t)o1; o[2] = (bf16_t)o2; o[3] = (bf16_t)o3;
  } else {
    ((float4*)out)[(size_t)row * (D_ / 4) + tid] = make_float4(o0, o1, o2, o3);
  }
}

// ---------------------------------------------------------------------------
// RoPE + permute [B,S,H,HD] -> [B,H,S,HD]  (bf16 in/out)
// ---------------------------------------------------------------------------
template <bool ROPE>
__global__ void permute_rope(const bf16_t* __restrict__ in,
                             const float* __restrict__ cosT,
                             const float* __restrict__ sinT,
                             bf16_t* __restrict__ out) {
  int t = blockIdx.x * blockDim.x + threadIdx.x;   // B*H*S*32 threads
  int i = t & 31;
  int r = t >> 5;
  int s = r % S_; r /= S_;
  int h = r % H_;
  int b = r / H_;
  size_t ib = ((size_t)(b * S_ + s) * H_ + h) * HD_;
  float v1 = (float)in[ib + i];
  float v2 = (float)in[ib + i + 32];
  float o1, o2;
  if (ROPE) {
    float c1 = cosT[s * HD_ + i],      s1 = sinT[s * HD_ + i];
    float c2 = cosT[s * HD_ + i + 32], s2 = sinT[s * HD_ + i + 32];
    o1 = v1 * c1 - v2 * s1;
    o2 = v2 * c2 + v1 * s2;
  } else {
    o1 = v1; o2 = v2;
  }
  size_t ob = ((size_t)(b * H_ + h) * S_ + s) * HD_;
  out[ob + i]      = (bf16_t)o1;
  out[ob + i + 32] = (bf16_t)o2;
}

// ---------------------------------------------------------------------------
// Tiled bf16 GEMM: C[M,N] = A[M,K] * B[K,N], WMMA 16x16x32 bf16, f32 accum.
// Block tile 128x128x32, 8 waves (4x2 wave grid, each wave 32x64).
// Double-buffered LDS staging; async global->LDS copies when available.
// ---------------------------------------------------------------------------
enum { EPI_BF16 = 0, EPI_SILU = 1, EPI_MUL = 2, EPI_ADDF32 = 3 };

#define GBM 128
#define GBN 128
#define GBK 32
#define GLDA 40    // padded A-tile stride (bf16 elems), 80B rows
#define GLDB 144   // padded B-tile stride (bf16 elems), 288B rows (32B aligned)

template <int EPI>
__global__ __launch_bounds__(256) void gemm_bf16(const bf16_t* __restrict__ A,
                                                 const bf16_t* __restrict__ Bm,
                                                 void* __restrict__ Dst,
                                                 const bf16_t* __restrict__ aux,
                                                 int M, int N, int K) {
  __shared__ __bf16 Als[2][GBM * GLDA];
  __shared__ __bf16 Bls[2][GBK * GLDB];
  int tid  = threadIdx.x;
  int lane = tid & 31, wave = tid >> 5;
  int wm = wave >> 1, wn = wave & 1;           // 4x2 wave grid (M x N)
  int half = lane >> 4, lm = lane & 15;
  int bm = blockIdx.y * GBM, bn = blockIdx.x * GBN;

  v8f acc[2][4];
#pragma unroll
  for (int i = 0; i < 2; ++i)
#pragma unroll
    for (int j = 0; j < 4; ++j) acc[i][j] = v8f_zero();

  // Stage one 128x32 A tile + 32x128 B tile into LDS buffer `buf`.
  auto stage = [&](int buf, int k0) {
    for (int i = tid; i < GBM * 4; i += 256) {
      int r = i >> 2, s = i & 3;
      cp_b128(&A[(size_t)(bm + r) * K + k0 + s * 8], &Als[buf][r * GLDA + s * 8]);
    }
    for (int i = tid; i < GBK * 16; i += 256) {
      int r = i >> 4, s = i & 15;
      cp_b128(&Bm[(size_t)(k0 + r) * N + bn + s * 8], &Bls[buf][r * GLDB + s * 8]);
    }
  };

  stage(0, 0);
  int nk = K / GBK;
  for (int it = 0; it < nk; ++it) {
    int buf = it & 1;
    async_wait_all();      // this wave's async deposits into Xls[buf] complete
    __syncthreads();       // all waves' deposits visible
    if (it + 1 < nk) stage(buf ^ 1, (it + 1) * GBK);  // overlap next tile

    int kb = half * 8;
    v16bf af[2];
#pragma unroll
    for (int mi = 0; mi < 2; ++mi) {
      int row = wm * 32 + mi * 16 + lm;        // A: lane = M row
      v8bf lo = *(const v8bf*)&Als[buf][row * GLDA + kb];
      v8bf hi = *(const v8bf*)&Als[buf][row * GLDA + 16 + kb];
      af[mi] = cat8(lo, hi);
    }
    v16bf bfg[4];
#pragma unroll
    for (int nj = 0; nj < 4; ++nj)             // B: lane = N col, elements over K (table order)
#pragma unroll
      for (int s = 0; s < 16; ++s)
        bfg[nj][s] = Bls[buf][(((s >> 3) << 4) + kb + (s & 7)) * GLDB + wn * 64 + nj * 16 + lm];
#pragma unroll
    for (int mi = 0; mi < 2; ++mi)
#pragma unroll
      for (int nj = 0; nj < 4; ++nj)
        acc[mi][nj] = wmma_bf16(af[mi], bfg[nj], acc[mi][nj]);
    // No trailing barrier: next iteration begins with asynccnt-wait + barrier,
    // and each wave consumed its LDS reads (into WMMA) before arriving there.
  }

  // Epilogue. C layout: VGPR j -> row (half*8 + j), col = lm.
#pragma unroll
  for (int mi = 0; mi < 2; ++mi)
#pragma unroll
    for (int nj = 0; nj < 4; ++nj) {
      int row0 = bm + wm * 32 + mi * 16 + half * 8;
      int col  = bn + wn * 64 + nj * 16 + lm;
#pragma unroll
      for (int j = 0; j < 8; ++j) {
        size_t idx = (size_t)(row0 + j) * N + col;
        float v = acc[mi][nj][j];
        if (EPI == EPI_BF16) {
          ((bf16_t*)Dst)[idx] = (bf16_t)v;
        } else if (EPI == EPI_SILU) {
          float s = v / (1.f + __expf(-v));
          ((bf16_t*)Dst)[idx] = (bf16_t)s;
        } else if (EPI == EPI_MUL) {
          float g = (float)aux[idx];
          ((bf16_t*)Dst)[idx] = (bf16_t)(v * g);
        } else {  // EPI_ADDF32: residual add into fp32 stream
          ((float*)Dst)[idx] += v;
        }
      }
    }
}

// ---------------------------------------------------------------------------
// WMMA flash attention. Grid (S/64, B*H), 128 threads (4 waves).
// Each wave owns 16 query rows; keys streamed in chunks of 32.
// Q/K/V are [B,H,S,HD] bf16 (RoPE pre-applied); Out is [B,S,D] bf16.
// ---------------------------------------------------------------------------
#define ALQ 72
#define ALKT 48
#define ALV 64
#define ALP 40

__global__ __launch_bounds__(128) void attn_kernel(const bf16_t* __restrict__ Q,
                                                   const bf16_t* __restrict__ Kx,
                                                   const bf16_t* __restrict__ V,
                                                   bf16_t* __restrict__ Out) {
  __shared__ __bf16 Qls[64 * ALQ];        // 64 q-rows x 64 hd
  __shared__ __bf16 Kls[64 * ALKT];       // K^T: [hd][key] 64x32
  __shared__ __bf16 Vls[32 * ALV];        // [key][hd] 32x64
  __shared__ __bf16 Pls[4 * 16 * ALP];    // per-wave 16x32 P transpose buffer

  int tid = threadIdx.x, lane = tid & 31, w = tid >> 5;
  int half = lane >> 4, lm = lane & 15;
  int qblk = blockIdx.x, bh = blockIdx.y;
  int b = bh / H_, h = bh % H_;
  int q0 = qblk * 64;
  const size_t base = (size_t)bh * S_ * HD_;

  // Load Q strip (64x64) into LDS.
  for (int i = tid; i < 64 * 8; i += 128) {
    int r = i >> 3, s = i & 7;
    *(v8bf*)&Qls[r * ALQ + s * 8] =
        *(const v8bf*)&Q[base + (size_t)(q0 + r) * HD_ + s * 8];
  }
  __syncthreads();

  int kb = half * 8;
  v16bf qa[2];
  {
    int row = w * 16 + lm;
#pragma unroll
    for (int c = 0; c < 2; ++c) {
      v8bf lo = *(const v8bf*)&Qls[row * ALQ + c * 32 + kb];
      v8bf hi = *(const v8bf*)&Qls[row * ALQ + c * 32 + 16 + kb];
      qa[c] = cat8(lo, hi);
    }
  }

  v8f o[4];
#pragma unroll
  for (int nj = 0; nj < 4; ++nj) o[nj] = v8f_zero();
  float mrow[8], lrow[8];
#pragma unroll
  for (int j = 0; j < 8; ++j) { mrow[j] = -1e30f; lrow[j] = 0.f; }

  int kend = q0 + 64;
  for (int kb0 = 0; kb0 < kend; kb0 += 32) {
    __syncthreads();  // previous chunk fully consumed
    // Load K chunk transposed (manual scatter) + V chunk (async copy).
    for (int i = tid; i < 32 * 8; i += 128) {
      int key = i >> 3, s = i & 7;
      v8bf kv = *(const v8bf*)&Kx[base + (size_t)(kb0 + key) * HD_ + s * 8];
#pragma unroll
      for (int e = 0; e < 8; ++e) Kls[(s * 8 + e) * ALKT + key] = kv[e];
      cp_b128(&V[base + (size_t)(kb0 + key) * HD_ + s * 8], &Vls[key * ALV + s * 8]);
    }
    async_wait_all();
    __syncthreads();

    // Scores: S = Q K^T (two 16x16 tiles, 2 WMMA each over HD=64).
    v8f sc[2];
#pragma unroll
    for (int t = 0; t < 2; ++t) {
      v8f c = v8f_zero();
      v16bf kf0, kf1;
#pragma unroll
      for (int s = 0; s < 16; ++s) {
        int kk = ((s >> 3) << 4) + kb + (s & 7);
        kf0[s] = Kls[(0 * 32 + kk) * ALKT + t * 16 + lm];
        kf1[s] = Kls[(1 * 32 + kk) * ALKT + t * 16 + lm];
      }
      c = wmma_bf16(qa[0], kf0, c);
      c = wmma_bf16(qa[1], kf1, c);
#pragma unroll
      for (int j = 0; j < 8; ++j) sc[t][j] = c[j] * 0.125f;  // HD^-0.5
    }
    // Causal mask.
    int rb = q0 + w * 16 + half * 8;
#pragma unroll
    for (int t = 0; t < 2; ++t)
#pragma unroll
      for (int j = 0; j < 8; ++j)
        if (kb0 + t * 16 + lm > rb + j) sc[t][j] = -1e30f;

    // Online softmax (rows live across 16 lanes of a half-wave).
    float mx[8];
#pragma unroll
    for (int j = 0; j < 8; ++j) mx[j] = fmaxf(sc[0][j], sc[1][j]);
    for (int off = 1; off < 16; off <<= 1)
#pragma unroll
      for (int j = 0; j < 8; ++j) mx[j] = fmaxf(mx[j], __shfl_xor(mx[j], off, 32));

    float ps[8];
#pragma unroll
    for (int j = 0; j < 8; ++j) {
      float mn = fmaxf(mrow[j], mx[j]);
      float alpha = __expf(mrow[j] - mn);
      mrow[j] = mn;
      float e0 = (sc[0][j] < -1e29f) ? 0.f : __expf(sc[0][j] - mn);
      float e1 = (sc[1][j] < -1e29f) ? 0.f : __expf(sc[1][j] - mn);
      sc[0][j] = e0; sc[1][j] = e1;
      ps[j] = e0 + e1;
      lrow[j] *= alpha;
#pragma unroll
      for (int nj = 0; nj < 4; ++nj) o[nj][j] *= alpha;
    }
    for (int off = 1; off < 16; off <<= 1)
#pragma unroll
      for (int j = 0; j < 8; ++j) ps[j] += __shfl_xor(ps[j], off, 32);
#pragma unroll
    for (int j = 0; j < 8; ++j) lrow[j] += ps[j];

    // Transpose P through per-wave LDS to A-matrix layout.
#pragma unroll
    for (int t = 0; t < 2; ++t)
#pragma unroll
      for (int j = 0; j < 8; ++j)
        Pls[(w * 16 + half * 8 + j) * ALP + t * 16 + lm] = (bf16_t)sc[t][j];
    v8bf plo = *(const v8bf*)&Pls[(w * 16 + lm) * ALP + kb];
    v8bf phi = *(const v8bf*)&Pls[(w * 16 + lm) * ALP + 16 + kb];
    v16bf pa = cat8(plo, phi);

    // O += P V  (4 output tiles over HD).
#pragma unroll
    for (int nj = 0; nj < 4; ++nj) {
      v16bf vf;
#pragma unroll
      for (int s = 0; s < 16; ++s)
        vf[s] = Vls[(((s >> 3) << 4) + kb + (s & 7)) * ALV + nj * 16 + lm];
      o[nj] = wmma_bf16(pa, vf, o[nj]);
    }
  }

  // Normalize and write [B,S,D] (head-major within row).
#pragma unroll
  for (int nj = 0; nj < 4; ++nj)
#pragma unroll
    for (int j = 0; j < 8; ++j) {
      int row = q0 + w * 16 + half * 8 + j;
      int col = nj * 16 + lm;
      float val = o[nj][j] / lrow[j];
      Out[((size_t)(b * S_ + row)) * D_ + h * HD_ + col] = (bf16_t)val;
    }
}

// ---------------------------------------------------------------------------
// Host orchestration
// ---------------------------------------------------------------------------
extern "C" void kernel_launch(void* const* d_in, const int* in_sizes, int n_in,
                              void* d_out, int out_size, void* d_ws, size_t ws_size,
                              hipStream_t stream) {
  (void)in_sizes; (void)n_in; (void)out_size; (void)ws_size;
  const float* x_processed = (const float*)d_in[0];
  const int*   boundaries  = (const int*)d_in[1];
  const int*   counts      = (const int*)d_in[2];
  const float* x_residual  = (const float*)d_in[3];
  const float* cosT        = (const float*)d_in[4];
  const float* sinT        = (const float*)d_in[5];
  const float* start_emb   = (const float*)d_in[7];
  const float* Wq = (const float*)d_in[8];
  const float* Wk = (const float*)d_in[9];
  const float* Wv = (const float*)d_in[10];
  const float* Wo = (const float*)d_in[11];
  const float* Wg = (const float*)d_in[12];
  const float* Wu = (const float*)d_in[13];
  const float* Wd = (const float*)d_in[14];
  const float* ln1 = (const float*)d_in[15];
  const float* ln2 = (const float*)d_in[16];
  const float* fnorm = (const float*)d_in[17];

  char* base = (char*)d_ws;
  size_t off = 0;
  auto alloc = [&](size_t bytes) -> void* {
    void* p = (void*)(base + off);
    off = (off + bytes + 255) & ~(size_t)255;
    return p;
  };

  const size_t DD = (size_t)D_ * D_;   // 1M
  const size_t DF = (size_t)D_ * F_;   // 4M
  bf16_t* wq_b = (bf16_t*)alloc(LAYERS * DD * 2);
  bf16_t* wk_b = (bf16_t*)alloc(LAYERS * DD * 2);
  bf16_t* wv_b = (bf16_t*)alloc(LAYERS * DD * 2);
  bf16_t* wo_b = (bf16_t*)alloc(LAYERS * DD * 2);
  bf16_t* wg_b = (bf16_t*)alloc(LAYERS * DF * 2);
  bf16_t* wu_b = (bf16_t*)alloc(LAYERS * DF * 2);
  bf16_t* wd_b = (bf16_t*)alloc(LAYERS * DF * 2);
  float*  x    = (float*)alloc((size_t)TOK * D_ * 4);
  bf16_t* hb   = (bf16_t*)alloc((size_t)TOK * D_ * 2);
  bf16_t* tmp  = (bf16_t*)alloc((size_t)TOK * D_ * 2);
  bf16_t* qh   = (bf16_t*)alloc((size_t)TOK * D_ * 2);
  bf16_t* kh   = (bf16_t*)alloc((size_t)TOK * D_ * 2);
  bf16_t* vh   = (bf16_t*)alloc((size_t)TOK * D_ * 2);
  bf16_t* gbuf = (bf16_t*)alloc((size_t)TOK * F_ * 2);
  bf16_t* act  = (bf16_t*)alloc((size_t)TOK * F_ * 2);
  int*    segb = (int*)alloc((size_t)TOK * 4);

  auto cvt = [&](const float* s, bf16_t* d, long long n) {
    cvt_f32_bf16<<<(unsigned)((n + 255) / 256), 256, 0, stream>>>(s, d, n);
  };
  cvt(Wq, wq_b, (long long)LAYERS * DD);
  cvt(Wk, wk_b, (long long)LAYERS * DD);
  cvt(Wv, wv_b, (long long)LAYERS * DD);
  cvt(Wo, wo_b, (long long)LAYERS * DD);
  cvt(Wg, wg_b, (long long)LAYERS * DF);
  cvt(Wu, wu_b, (long long)LAYERS * DF);
  cvt(Wd, wd_b, (long long)LAYERS * DF);

  seg_kernel<<<(TOK + 255) / 256, 256, 0, stream>>>(boundaries, counts, segb);
  gather_kernel<<<(TOK * (D_ / 4) + 255) / 256, 256, 0, stream>>>(
      x_processed, x_residual, start_emb, segb, x);

  const int PR_BLOCKS = (B_ * H_ * S_ * 32) / 256;
  dim3 g1(D_ / GBN, TOK / GBM);   // N=1024 GEMMs
  dim3 g2(F_ / GBN, TOK / GBM);   // N=4096 GEMMs
  dim3 ga(S_ / 64, B_ * H_);

  for (int l = 0; l < LAYERS; ++l) {
    rmsnorm_k<true><<<TOK, 256, 0, stream>>>(x, ln1 + (size_t)l * D_, hb);

    gemm_bf16<EPI_BF16><<<g1, 256, 0, stream>>>(hb, wq_b + (size_t)l * DD, tmp,
                                                nullptr, TOK, D_, D_);
    permute_rope<true><<<PR_BLOCKS, 256, 0, stream>>>(tmp, cosT, sinT, qh);
    gemm_bf16<EPI_BF16><<<g1, 256, 0, stream>>>(hb, wk_b + (size_t)l * DD, tmp,
                                                nullptr, TOK, D_, D_);
    permute_rope<true><<<PR_BLOCKS, 256, 0, stream>>>(tmp, cosT, sinT, kh);
    gemm_bf16<EPI_BF16><<<g1, 256, 0, stream>>>(hb, wv_b + (size_t)l * DD, tmp,
                                                nullptr, TOK, D_, D_);
    permute_rope<false><<<PR_BLOCKS, 256, 0, stream>>>(tmp, cosT, sinT, vh);

    attn_kernel<<<ga, 128, 0, stream>>>(qh, kh, vh, tmp);
    gemm_bf16<EPI_ADDF32><<<g1, 256, 0, stream>>>(tmp, wo_b + (size_t)l * DD, x,
                                                  nullptr, TOK, D_, D_);

    rmsnorm_k<true><<<TOK, 256, 0, stream>>>(x, ln2 + (size_t)l * D_, hb);
    gemm_bf16<EPI_SILU><<<g2, 256, 0, stream>>>(hb, wg_b + (size_t)l * DF, gbuf,
                                                nullptr, TOK, F_, D_);
    gemm_bf16<EPI_MUL><<<g2, 256, 0, stream>>>(hb, wu_b + (size_t)l * DF, act,
                                               gbuf, TOK, F_, D_);
    gemm_bf16<EPI_ADDF32><<<g1, 256, 0, stream>>>(act, wd_b + (size_t)l * DF, x,
                                                  nullptr, TOK, D_, F_);
  }
  rmsnorm_k<false><<<TOK, 256, 0, stream>>>(x, fnorm, d_out);
}
